// SpatialAttention_2061584302176
// MI455X (gfx1250) — matrix operand order repaired and checked by reading, not compiled, not run on hardware
//
#include <hip/hip_runtime.h>
#include <hip/hip_bf16.h>

typedef __attribute__((ext_vector_type(16))) __bf16 v16bf;
typedef __attribute__((ext_vector_type(8)))  __bf16 v8bf;
typedef __attribute__((ext_vector_type(4)))  __bf16 v4bf;
typedef __attribute__((ext_vector_type(8)))  float  v8f;
typedef __attribute__((ext_vector_type(4)))  float  v4f;

#define BT      2048
#define N_TOK   64
#define D_DIM   256
#define SCALE   0.0625f     // 1/sqrt(256)
#define BCOEF   0.01f

// Padded LDS strides (elements) -> row-to-row bank offsets != 0 mod 64 dwords
#define XQK_STRIDE  264     // bf16, 528 B/row (132 dwords, +4 banks/row)
#define VT_STRIDE   72      // bf16, 144 B/row (36 dwords/row)
#define S_STRIDE    68      // f32,  272 B/row
#define P_STRIDE    72      // bf16, 144 B/row

// LDS carve (bytes). S and P overlay X (dead after QKV stage, barriers between).
#define OFF_X    0
#define OFF_S    0
#define OFF_P    17408                    // 64*68*4, 16B aligned
#define OFF_Q    33792                    // 64*264*2
#define OFF_K    67584
#define OFF_V    101376
#define OFF_POS  138240                   // OFF_V + 256*72*2
#define OFF_RA   138752
#define OFF_RB   139776
#define SMEM_BYTES 140800

// ---- WMMA fragment loaders (CDNA5 16-bit layouts, CDNA5 ISA) ----

// A (16x32 MxK): lane holds row M = lane&15; lanes 0-15: K 0..7 & 16..23,
// lanes 16-31: K 8..15 & 24..31.
__device__ __forceinline__ v16bf load_fragA(const __bf16* rowPtr, int k0, int lane) {
    const int kb = k0 + ((lane >> 4) << 3);
    union { v16bf v; v8bf h[2]; } u;
    u.h[0] = *(const v8bf*)(rowPtr + kb);        // K = kb .. kb+7
    u.h[1] = *(const v8bf*)(rowPtr + kb + 16);   // K = kb+16 .. kb+23
    return u.v;
}

// B (32x16 KxN): lane holds column N = lane&15; lanes 0-15: K 0..7 & 16..23,
// lanes 16-31: K 8..15 & 24..31 (matches A). rowPtr = &Bstored[n][0],
// Bstored being the [N][K] row-major (i.e. B-transposed) array.
// Two 16B loads: safe for 16B-aligned (not 32B) padded rows.
__device__ __forceinline__ v16bf load_fragB(const __bf16* rowPtr, int k0, int lane) {
    const int kb = k0 + ((lane >> 4) << 3);
    union { v16bf v; v8bf h[2]; } u;
    u.h[0] = *(const v8bf*)(rowPtr + kb);
    u.h[1] = *(const v8bf*)(rowPtr + kb + 16);
    return u.v;
}

__device__ __forceinline__ v8f wmma_bf16(v16bf a, v16bf b, v8f c) {
    return __builtin_amdgcn_wmma_f32_16x16x32_bf16(false, a, false, b,
                                                   (short)0, c, false, false);
}

// ---- prep: Wq/Wk/Wv f32 [256][256] -> bf16 workspace ----
__global__ void prep_weights_bf16(const float* __restrict__ Wq,
                                  const float* __restrict__ Wk,
                                  const float* __restrict__ Wv,
                                  __bf16* __restrict__ wbf) {
    const int i = blockIdx.x * blockDim.x + threadIdx.x;   // 0 .. 3*65536-1
    const float* src = (i < 65536) ? Wq : ((i < 131072) ? Wk : Wv);
    wbf[i] = (__bf16)src[i & 65535];
}

// ---- main: one block (8 waves) per batch ----
__launch_bounds__(256)
__global__ void spatial_attn_kernel(const float* __restrict__ features,
                                    const float* __restrict__ positions,
                                    const __bf16* __restrict__ Wbf,   // [3][256][256]
                                    const float* __restrict__ bq,
                                    const float* __restrict__ bk,
                                    const float* __restrict__ bv,
                                    float* __restrict__ out) {
    __shared__ __align__(16) unsigned char smem[SMEM_BYTES];
    __bf16* Xs  = (__bf16*)(smem + OFF_X);   // [64][264]
    float*  Ss  = (float*) (smem + OFF_S);   // [64][68]  (overlays Xs)
    __bf16* Ps  = (__bf16*)(smem + OFF_P);   // [64][72]  (overlays Xs)
    __bf16* Qs  = (__bf16*)(smem + OFF_Q);   // [64][264]
    __bf16* Ks  = (__bf16*)(smem + OFF_K);   // [64][264]
    __bf16* Vt  = (__bf16*)(smem + OFF_V);   // [256][72] transposed V
    float*  Pos = (float*) (smem + OFF_POS); // [64][2]
    float*  RedA= (float*) (smem + OFF_RA);  // [256]
    float*  RedB= (float*) (smem + OFF_RB);  // [256]

    const int tid  = threadIdx.x;
    const int lane = tid & 31;
    const int wave = tid >> 5;
    const int col  = lane & 15;            // tile column owned by this lane (C/D)
    const int rowoff = (lane >> 4) << 3;   // lanes>=16 hold rows M+8
    const long fbase = (long)blockIdx.x * (N_TOK * D_DIM);

    // ---------- Stage A: X (f32, streamed once) -> bf16 LDS; positions ----------
    for (int v = tid; v < (N_TOK * D_DIM) / 4; v += 256) {
        const int r = v >> 6;              // row 0..63
        const int c = (v & 63) * 4;        // col 0..252
        const v4f f = __builtin_nontemporal_load(
            (const v4f*)(features + fbase + (long)r * D_DIM + c));
        v4bf b;
        b[0] = (__bf16)f[0]; b[1] = (__bf16)f[1];
        b[2] = (__bf16)f[2]; b[3] = (__bf16)f[3];
        *(v4bf*)&Xs[r * XQK_STRIDE + c] = b;
    }
    if (tid < N_TOK * 2)
        Pos[tid] = positions[(long)blockIdx.x * (N_TOK * 2) + tid];
    __syncthreads();

    // ---------- Stage B: Q/K/V = X @ W^T + b ----------
    // Wave w owns output-column tiles {2w, 2w+1}. W rows [d_out][c] are exactly
    // the per-lane-contiguous B-operand layout. B fragment reused across 4 M-tiles;
    // mat loop not unrolled to bound register pressure.
    for (int i = 0; i < 2; ++i) {
        const int d0 = (2 * wave + i) * 16;
        #pragma unroll 1
        for (int mat = 0; mat < 3; ++mat) {
            const __bf16* Wrow = Wbf + mat * (D_DIM * D_DIM) + (d0 + col) * D_DIM;
            const float* bp = (mat == 0) ? bq : (mat == 1) ? bk : bv;
            const float bias = bp[d0 + col];
            v8f acc[4];
            #pragma unroll
            for (int mt = 0; mt < 4; ++mt)
                #pragma unroll
                for (int j = 0; j < 8; ++j) acc[mt][j] = 0.0f;
            #pragma unroll
            for (int k0 = 0; k0 < D_DIM; k0 += 32) {
                const v16bf b = load_fragB(Wrow, k0, lane);
                #pragma unroll
                for (int mt = 0; mt < 4; ++mt) {
                    const v16bf a =
                        load_fragA(Xs + (mt * 16 + col) * XQK_STRIDE, k0, lane);
                    acc[mt] = wmma_bf16(a, b, acc[mt]);
                }
            }
            if (mat == 2) {
                // V: transposed store, 8 consecutive m per lane -> one b128
                #pragma unroll
                for (int mt = 0; mt < 4; ++mt) {
                    v8bf pk;
                    #pragma unroll
                    for (int j = 0; j < 8; ++j) pk[j] = (__bf16)(acc[mt][j] + bias);
                    *(v8bf*)&Vt[(d0 + col) * VT_STRIDE + mt * 16 + rowoff] = pk;
                }
            } else {
                __bf16* dst = (mat == 0) ? Qs : Ks;
                #pragma unroll
                for (int mt = 0; mt < 4; ++mt)
                    #pragma unroll
                    for (int j = 0; j < 8; ++j)
                        dst[(mt * 16 + rowoff + j) * XQK_STRIDE + d0 + col] =
                            (__bf16)(acc[mt][j] + bias);
            }
        }
    }
    __syncthreads();

    // ---------- Stage C: S = (Q @ K^T) * scale ----------
    for (int i = 0; i < 2; ++i) {
        const int t  = 2 * wave + i;
        const int n0 = (t >> 2) * 16;
        const int m0 = (t & 3) * 16;
        const __bf16* arow = Qs + (n0 + col) * XQK_STRIDE;
        const __bf16* brow = Ks + (m0 + col) * XQK_STRIDE;
        v8f acc;
        #pragma unroll
        for (int j = 0; j < 8; ++j) acc[j] = 0.0f;
        #pragma unroll
        for (int k0 = 0; k0 < D_DIM; k0 += 32)
            acc = wmma_bf16(load_fragA(arow, k0, lane),
                            load_fragB(brow, k0, lane), acc);
        #pragma unroll
        for (int j = 0; j < 8; ++j)
            Ss[(n0 + rowoff + j) * S_STRIDE + m0 + col] = acc[j] * SCALE;
    }
    __syncthreads();

    // ---------- Stage D: spatial bias + softmax (4 threads per row) ----------
    {
        const int row = tid >> 2;
        const int q4  = (tid & 3) * 16;
        const float px = Pos[row * 2 + 0];
        const float py = Pos[row * 2 + 1];
        float* srow = Ss + row * S_STRIDE;
        float mx = -3.402823466e38f;
        for (int m = q4; m < q4 + 16; ++m) {
            const float dx = px - Pos[m * 2 + 0];
            const float dy = py - Pos[m * 2 + 1];
            const float s  = srow[m] - BCOEF * sqrtf(dx * dx + dy * dy);
            srow[m] = s;
            mx = fmaxf(mx, s);
        }
        RedA[tid] = mx;
        __syncthreads();
        mx = fmaxf(fmaxf(RedA[row * 4 + 0], RedA[row * 4 + 1]),
                   fmaxf(RedA[row * 4 + 2], RedA[row * 4 + 3]));
        float sum = 0.0f;
        for (int m = q4; m < q4 + 16; ++m) {
            const float e = __expf(srow[m] - mx);
            srow[m] = e;
            sum += e;
        }
        RedB[tid] = sum;
        __syncthreads();
        sum = (RedB[row * 4 + 0] + RedB[row * 4 + 1]) +
              (RedB[row * 4 + 2] + RedB[row * 4 + 3]);
        const float inv = 1.0f / sum;
        __bf16* prow = Ps + row * P_STRIDE;
        for (int m = q4; m < q4 + 16; ++m)
            prow[m] = (__bf16)(srow[m] * inv);
    }
    __syncthreads();

    // ---------- Stage E: O = P @ V (B operand from transposed Vt) ----------
    float* obase = out + fbase;
    for (int i = 0; i < 2; ++i) {
        const int d0 = (2 * wave + i) * 16;
        const __bf16* brow = Vt + (d0 + col) * VT_STRIDE;  // Vt[d][*] == V[*][d]
        #pragma unroll 1
        for (int mt = 0; mt < 4; ++mt) {
            const int n0 = mt * 16;
            const __bf16* arow = Ps + (n0 + col) * P_STRIDE;
            v8f acc;
            #pragma unroll
            for (int j = 0; j < 8; ++j) acc[j] = 0.0f;
            #pragma unroll
            for (int k0 = 0; k0 < N_TOK; k0 += 32)
                acc = wmma_bf16(load_fragA(arow, k0, lane),
                                load_fragB(brow, k0, lane), acc);
            #pragma unroll
            for (int j = 0; j < 8; ++j)
                __builtin_nontemporal_store(
                    acc[j],
                    obase + (long)(n0 + rowoff + j) * D_DIM + d0 + col);
        }
    }
}

extern "C" void kernel_launch(void* const* d_in, const int* in_sizes, int n_in,
                              void* d_out, int out_size, void* d_ws, size_t ws_size,
                              hipStream_t stream) {
    const float* features  = (const float*)d_in[0];
    const float* positions = (const float*)d_in[1];
    const float* Wq = (const float*)d_in[2];
    const float* bq = (const float*)d_in[3];
    const float* Wk = (const float*)d_in[4];
    const float* bk = (const float*)d_in[5];
    const float* Wv = (const float*)d_in[6];
    const float* bv = (const float*)d_in[7];
    float* outp = (float*)d_out;
    __bf16* wbf = (__bf16*)d_ws;   // needs 3*256*256*2 = 384 KB workspace

    prep_weights_bf16<<<768, 256, 0, stream>>>(Wq, Wk, Wv, wbf);
    spatial_attn_kernel<<<BT, 256, 0, stream>>>(features, positions, wbf,
                                                bq, bk, bv, outp);
}
